// MixtureOfHeadsAttention_20985210208477
// MI455X (gfx1250) — compile-verified, matrix operand order checked by reading, not run on hardware
//
#include <hip/hip_runtime.h>
#include <hip/hip_bf16.h>
#include <math.h>

// ---------------- constants ----------------
#define Hh 32
#define Kk 8
#define Dd 128
#define Ee 1024
#define Ss 1024
#define SK (Ss * Kk)          // 8192
#define HD (Hh * Dd)          // 4096
#define MAX_SEQ 8192
#define SCALE 0.08838834764831845f   // 128^-0.5
#define LB_W 0.01f
#define ENT_W 0.01f

typedef unsigned short u16;
typedef __attribute__((ext_vector_type(16))) __bf16 bf16x16;
typedef __attribute__((ext_vector_type(8)))  float  f32x8;
typedef __attribute__((ext_vector_type(8)))  u16    u16x8;
typedef __attribute__((ext_vector_type(16))) u16    u16x16;

union fragu { u16x16 u; bf16x16 b; };

__device__ __forceinline__ u16 f2bf(float f) {
  unsigned u = __builtin_bit_cast(unsigned, f);
  unsigned r = u + 0x7FFFu + ((u >> 16) & 1u);
  return (u16)(r >> 16);
}

// Load a 16xK-chunk WMMA operand (A row-major, or B from a transposed
// "Bt" array where Bt[n][k]). ISA 7.12.2 bf16 layout:
// lane l: row m = l&15 ; elements 0..7 -> k = kc + 8*(l>>4) + e
//                        elements 8..15 -> k = kc + 16 + 8*(l>>4) + (e-8)
__device__ __forceinline__ bf16x16 load_frag(const u16* base, int row_stride,
                                             int row0, int kc, int lane) {
  int m = row0 + (lane & 15);
  int kh = (lane >> 4) << 3;
  const u16* p0 = base + (long)m * row_stride + kc + kh;
  u16x8 lo = *(const u16x8*)p0;
  u16x8 hi = *(const u16x8*)(p0 + 16);
  fragu f;
#pragma unroll
  for (int i = 0; i < 8; ++i) { f.u[i] = lo[i]; f.u[i + 8] = hi[i]; }
  return f.b;
}

__device__ __forceinline__ f32x8 wmma_bf16(bf16x16 a, bf16x16 b, f32x8 c) {
  return __builtin_amdgcn_wmma_f32_16x16x32_bf16(false, a, false, b,
                                                 (short)0, c, false, false);
}

// CDNA5 async copy: global -> LDS, 16 bytes per enabled lane, ASYNCcnt-tracked.
__device__ __forceinline__ void async_b128(unsigned lds_off, const u16* gsrc) {
  asm volatile("global_load_async_to_lds_b128 %0, %1, off"
               :: "v"(lds_off), "v"(gsrc) : "memory");
}
__device__ __forceinline__ void wait_async0() {
  asm volatile("s_wait_asynccnt 0x0" ::: "memory");
}

// ---------------- tiny utility kernels ----------------
__global__ void k_zero_f32(float* p, int n) {
  int i = blockIdx.x * blockDim.x + threadIdx.x;
  if (i < n) p[i] = 0.f;
}

__global__ void k_cvt_bf16(const float* __restrict__ in, u16* __restrict__ out, int n) {
  int i = blockIdx.x * blockDim.x + threadIdx.x;
  if (i < n) out[i] = f2bf(in[i]);
}

// W (Kd x N) fp32 row-major -> Wt (N x Kd) bf16 row-major
__global__ void k_cvt_transpose(const float* __restrict__ W, u16* __restrict__ Wt,
                                int Kd, int N) {
  int i = blockIdx.x * blockDim.x + threadIdx.x;
  if (i >= Kd * N) return;
  int k = i / N, n = i - k * N;
  Wt[(long)n * Kd + k] = f2bf(W[i]);
}

// ---------------- router: logits, softmax, top-8, aux partials ----------------
__global__ void k_router(const float* __restrict__ x, const float* __restrict__ Wr,
                         int* __restrict__ head_idx, float* __restrict__ head_w,
                         float* __restrict__ F_sum, float* __restrict__ P_sum,
                         float* __restrict__ ent_z) {
  int wave = threadIdx.x >> 5;
  int lane = threadIdx.x & 31;
  int s = blockIdx.x * (blockDim.x >> 5) + wave;
  if (s >= Ss) return;

  // logit for head = lane
  float acc = 0.f;
  const float* xr = x + (long)s * Ee;
  for (int e = 0; e < Ee; ++e) acc = fmaf(xr[e], Wr[e * Hh + lane], acc);
  float logit = acc;

  // full softmax over 32 heads (one per lane)
  float mx = logit;
#pragma unroll
  for (int off = 16; off >= 1; off >>= 1) mx = fmaxf(mx, __shfl_xor(mx, off));
  float ex = __expf(logit - mx);
  float sum = ex;
#pragma unroll
  for (int off = 16; off >= 1; off >>= 1) sum += __shfl_xor(sum, off);
  float prob = ex / sum;

  // entropy & z-loss partials
  float entp = -(prob * __logf(prob + 1e-8f));
#pragma unroll
  for (int off = 16; off >= 1; off >>= 1) entp += __shfl_xor(entp, off);
  float lse = mx + __logf(sum);

  // iterative top-8 (ties -> lowest index, like jax top_k)
  float rem = logit;
  float myv = 0.f; int myi = 0; float mx8 = 0.f; float sum8 = 0.f;
  int primary = 0;
#pragma unroll
  for (int k = 0; k < Kk; ++k) {
    float v = rem; int idx = lane;
#pragma unroll
    for (int off = 16; off >= 1; off >>= 1) {
      float ov = __shfl_xor(v, off); int oi = __shfl_xor(idx, off);
      if (ov > v || (ov == v && oi < idx)) { v = ov; idx = oi; }
    }
    if (k == 0) { mx8 = v; primary = idx; }
    sum8 += __expf(v - mx8);
    if (lane == k) { myv = v; myi = idx; }
    if (lane == idx) rem = -1e30f;
  }

  if (lane < Kk) {
    head_idx[s * Kk + lane] = myi;
    head_w[s * Kk + lane] = __expf(myv - mx8) / sum8;
  }
  atomicAdd(&P_sum[lane], prob);
  if (lane == 0) {
    atomicAdd(&F_sum[primary], 1.f);
    atomicAdd(&ent_z[0], entp);
    atomicAdd(&ent_z[1], lse * lse);
  }
}

// ---------------- sequential rank/positions + counts + aux ----------------
__global__ void k_rank_aux(const int* __restrict__ head_idx,
                           const int* __restrict__ counts_in,
                           int* __restrict__ positions,
                           int* __restrict__ counts_out,
                           const float* __restrict__ F_sum,
                           const float* __restrict__ P_sum,
                           const float* __restrict__ ent_z,
                           float* __restrict__ aux_out) {
  int lane = threadIdx.x;   // one head per lane, 32 lanes
  int cnt = 0;
  int base = counts_in[lane];
  for (int i = 0; i < SK; ++i) {
    int h = head_idx[i];
    if (h == lane) {
      int p = cnt + base;
      positions[i] = p < (MAX_SEQ - 1) ? (p < 0 ? 0 : p) : (MAX_SEQ - 1);
      ++cnt;
    }
  }
  counts_out[lane] = base + cnt;
  if (lane == 0) {
    float bal = 0.f;
    for (int h = 0; h < Hh; ++h)
      bal += (F_sum[h] / (float)Ss) * (P_sum[h] / (float)Ss);
    bal *= (float)Hh;
    float ent_mean = ent_z[0] / (float)Ss;
    float z_mean = ent_z[1] / (float)Ss;
    aux_out[0] = LB_W * bal + ENT_W * (-ent_mean) + 0.01f * z_mean;
  }
}

// ------------- bf16 WMMA GEMM, 16x64 per wave, 2-stage pipelined -------------
// C = A(MxKd) * Bt(NxKd)^T
__global__ void k_gemm_bf16(const u16* __restrict__ A, const u16* __restrict__ Bt,
                            float* __restrict__ C, int M, int N, int Kd) {
  int wave = threadIdx.x >> 5;
  int lane = threadIdx.x & 31;
  int n0 = (blockIdx.x * 4 + wave) * 64;    // 64-col macro tile per wave
  int m0 = blockIdx.y * 16;
  if (n0 >= N || m0 >= M) return;

  f32x8 acc[4] = {};
  // stage 0 preload
  bf16x16 a = load_frag(A, Kd, m0, 0, lane);
  bf16x16 b[4];
#pragma unroll
  for (int j = 0; j < 4; ++j) b[j] = load_frag(Bt, Kd, n0 + j * 16, 0, lane);

  for (int kc = 0; kc < Kd; kc += 32) {
    int kn = (kc + 32 < Kd) ? kc + 32 : kc;   // next chunk (dummy on last iter)
    __builtin_prefetch(A + (long)(m0 + (lane & 15)) * Kd + kn + 32, 0, 1);
    __builtin_prefetch(Bt + (long)(n0 + (lane & 15)) * Kd + kn + 32, 0, 1);
    // issue next-stage loads before consuming current frags -> VMEM/WMMA overlap
    bf16x16 an = load_frag(A, Kd, m0, kn, lane);
    bf16x16 bn[4];
#pragma unroll
    for (int j = 0; j < 4; ++j) bn[j] = load_frag(Bt, Kd, n0 + j * 16, kn, lane);
#pragma unroll
    for (int j = 0; j < 4; ++j) acc[j] = wmma_bf16(a, b[j], acc[j]);
    a = an;
#pragma unroll
    for (int j = 0; j < 4; ++j) b[j] = bn[j];
  }
  // C layout: vgpr g, lane l -> row = g + 8*(l>>4), col = l&15
  int rbase = m0 + ((lane >> 4) << 3);
#pragma unroll
  for (int j = 0; j < 4; ++j) {
    int col = n0 + j * 16 + (lane & 15);
#pragma unroll
    for (int g = 0; g < 8; ++g) C[(long)(rbase + g) * N + col] = acc[j][g];
  }
}

// ---------------- gather selected heads + RoPE; V stored transposed ----------------
__global__ void k_gather_rope(const float* __restrict__ q_all,
                              const float* __restrict__ k_all,
                              const float* __restrict__ v_all,
                              const int* __restrict__ head_idx,
                              const int* __restrict__ positions,
                              u16* __restrict__ Qb, u16* __restrict__ Kb,
                              u16* __restrict__ Vt) {
  int wave = threadIdx.x >> 5;
  int lane = threadIdx.x & 31;
  int i = blockIdx.x * (blockDim.x >> 5) + wave;   // flat (s,k) row
  if (i >= SK) return;
  int s = i >> 3;
  int h = head_idx[i];
  float pos = (float)positions[i];
  long base = (long)s * HD + (long)h * Dd;
  const float L10K = 9.210340371976184f;  // ln(10000)
#pragma unroll
  for (int c = 0; c < 4; ++c) {
    int d = c * 32 + lane;
    int d2 = d & 63;
    float inv = __expf(-((float)(2 * d2) / (float)Dd) * L10K);
    float fr = pos * inv;
    float co = __cosf(fr), si = __sinf(fr);
    float qv = q_all[base + d], kv = k_all[base + d], vv = v_all[base + d];
    float qrot = (d < 64) ? -q_all[base + d + 64] : q_all[base + d - 64];
    float krot = (d < 64) ? -k_all[base + d + 64] : k_all[base + d - 64];
    Qb[(long)i * Dd + d] = f2bf(qv * co + qrot * si);
    Kb[(long)i * Dd + d] = f2bf(kv * co + krot * si);
    Vt[(long)d * SK + i] = f2bf(vv);   // transposed: d-major
  }
}

// ---------------- flash attention over gathered rows ----------------
// Block = 4 waves = 4 consecutive 16-row q tiles. K/V key-blocks shared by all
// waves are staged into LDS with CDNA5 async global->LDS copies (ASYNCcnt).
__global__ void k_flash(const u16* __restrict__ Qb, const u16* __restrict__ Kb,
                        const u16* __restrict__ Vt,
                        const int* __restrict__ head_flat,
                        const float* __restrict__ head_w_flat,
                        float* __restrict__ Out) {
  __shared__ __align__(16) u16 kst[32 * Dd];      // K block: 32 keys x 128d (8KB)
  __shared__ __align__(16) u16 vst[Dd * 32];      // Vt block: 128d x 32 keys (8KB)
  __shared__ __align__(16) u16 psh[4][16 * 32];   // per-wave P tile (4KB)
  int tid = threadIdx.x;
  int wave = tid >> 5;
  int lane = tid & 31;
  int t = blockIdx.x * 4 + wave;          // 16-row q tile, t in [0,512)
  int qbase = t * 16;
  int halfoff = (lane >> 4) << 3;
  int col = lane & 15;
  u16* pw = &psh[wave][0];
  unsigned lk = (unsigned)(size_t)&kst[0];
  unsigned lv = (unsigned)(size_t)&vst[0];

  bf16x16 qa[4];
#pragma unroll
  for (int c = 0; c < 4; ++c) qa[c] = load_frag(Qb, Dd, qbase, c * 32, lane);

  int hq[8];
#pragma unroll
  for (int g = 0; g < 8; ++g) hq[g] = head_flat[qbase + g + halfoff];

  f32x8 acc[8] = {};
  float mstat[8], lstat[8];
#pragma unroll
  for (int g = 0; g < 8; ++g) { mstat[g] = -1e30f; lstat[g] = 0.f; }

  int my_nblk = (t >> 1) + 1;                         // causal for this wave
  int max_nblk = ((blockIdx.x * 4 + 3) >> 1) + 1;     // union over block

  for (int b = 0; b < max_nblk; ++b) {
    int key0 = b * 32;
    __syncthreads();   // previous iteration's LDS readers done
    // ---- cooperative async stage: K block (contiguous 8KB) ----
    {
      const u16* gk = Kb + (long)key0 * Dd;
#pragma unroll
      for (int i = 0; i < 4; ++i) {
        int o = tid * 8 + i * 1024;                   // u16 units, 16B chunks
        async_b128(lk + (unsigned)(o * 2), gk + o);
      }
      // ---- Vt block: 128 rows (d) x 32 keys, row stride SK ----
#pragma unroll
      for (int i = 0; i < 4; ++i) {
        int c = tid + i * 128;                        // 512 chunks of 16B
        int d = c >> 2, sub = c & 3;
        async_b128(lv + (unsigned)((d * 32 + sub * 8) * 2),
                   Vt + (long)d * SK + key0 + sub * 8);
      }
    }
    wait_async0();
    __syncthreads();   // staged data visible to all waves

    if (b < my_nblk) {
      // batch all 8 K-fragment DS loads, then the 8-WMMA QK burst
      bf16x16 bk[8];
#pragma unroll
      for (int c = 0; c < 4; ++c) {
        bk[2 * c]     = load_frag(kst, Dd, 0, c * 32, lane);    // keys 0-15
        bk[2 * c + 1] = load_frag(kst, Dd, 16, c * 32, lane);   // keys 16-31
      }
      f32x8 s0 = {}, s1 = {};
#pragma unroll
      for (int c = 0; c < 4; ++c) {
        s0 = wmma_bf16(qa[c], bk[2 * c], s0);
        s1 = wmma_bf16(qa[c], bk[2 * c + 1], s1);
      }
      int kj0 = key0 + col, kj1 = kj0 + 16;
      int hk0 = head_flat[kj0], hk1 = head_flat[kj1];
#pragma unroll
      for (int g = 0; g < 8; ++g) {
        int qi = qbase + g + halfoff;
        float v0 = (hk0 == hq[g] && kj0 <= qi) ? s0[g] * SCALE : -1e30f;
        float v1 = (hk1 == hq[g] && kj1 <= qi) ? s1[g] * SCALE : -1e30f;
        float rmax = fmaxf(v0, v1);
#pragma unroll
        for (int off = 8; off >= 1; off >>= 1) rmax = fmaxf(rmax, __shfl_xor(rmax, off));
        float mnew = fmaxf(mstat[g], rmax);
        float corr = __expf(mstat[g] - mnew);
        float p0 = __expf(v0 - mnew), p1 = __expf(v1 - mnew);
        float rsum = p0 + p1;
#pragma unroll
        for (int off = 8; off >= 1; off >>= 1) rsum += __shfl_xor(rsum, off);
        lstat[g] = lstat[g] * corr + rsum;
        mstat[g] = mnew;
#pragma unroll
        for (int cb = 0; cb < 8; ++cb) acc[cb][g] *= corr;
        int r = g + halfoff;
        pw[r * 32 + col] = f2bf(p0);
        pw[r * 32 + 16 + col] = f2bf(p1);
      }
      asm volatile("s_wait_dscnt 0x0" ::: "memory");
      bf16x16 pa = load_frag(pw, 32, 0, 0, lane);   // P as A-operand (16x32)
      // batch all 8 V-fragment DS loads, then the 8-WMMA PV burst
      bf16x16 bv[8];
#pragma unroll
      for (int cb = 0; cb < 8; ++cb) bv[cb] = load_frag(vst, 32, cb * 16, 0, lane);
#pragma unroll
      for (int cb = 0; cb < 8; ++cb) acc[cb] = wmma_bf16(pa, bv[cb], acc[cb]);
    }
  }

#pragma unroll
  for (int g = 0; g < 8; ++g) {
    int qi = qbase + g + halfoff;
    float w = head_w_flat[qi] / lstat[g];
#pragma unroll
    for (int cb = 0; cb < 8; ++cb)
      Out[(long)qi * Dd + cb * 16 + col] = acc[cb][g] * w;
  }
}

// ---------------- launcher ----------------
extern "C" void kernel_launch(void* const* d_in, const int* in_sizes, int n_in,
                              void* d_out, int out_size, void* d_ws, size_t ws_size,
                              hipStream_t stream) {
  const float* x  = (const float*)d_in[0];
  const float* Wq = (const float*)d_in[1];
  const float* Wk = (const float*)d_in[2];
  const float* Wv = (const float*)d_in[3];
  const float* Wr = (const float*)d_in[4];
  const float* Wo = (const float*)d_in[5];
  const int* counts_in = (const int*)d_in[6];

  float* out_f = (float*)d_out;                       // (1,S,E)
  int*   counts_out = (int*)((float*)d_out + Ss * Ee);
  float* aux_out = (float*)d_out + Ss * Ee + Hh;

  // workspace carve-up (256B aligned)
  char* ws = (char*)d_ws;
  size_t off = 0;
  auto alloc = [&](size_t bytes) { void* p = ws + off; off = (off + bytes + 255) & ~(size_t)255; return p; };
  u16* xb   = (u16*)alloc((size_t)Ss * Ee * 2);
  u16* WqT  = (u16*)alloc((size_t)HD * Ee * 2);
  u16* WkT  = (u16*)alloc((size_t)HD * Ee * 2);
  u16* WvT  = (u16*)alloc((size_t)HD * Ee * 2);
  u16* WoT  = (u16*)alloc((size_t)Ee * (Kk * Dd) * 2);
  float* q_all = (float*)alloc((size_t)Ss * HD * 4);
  float* k_all = (float*)alloc((size_t)Ss * HD * 4);
  float* v_all = (float*)alloc((size_t)Ss * HD * 4);
  int*   head_idx = (int*)alloc(SK * 4);
  float* head_w   = (float*)alloc(SK * 4);
  int*   positions = (int*)alloc(SK * 4);
  float* F_sum = (float*)alloc(Hh * 4);
  float* P_sum = (float*)alloc(Hh * 4);
  float* ent_z = (float*)alloc(2 * 4);
  u16* Qb = (u16*)alloc((size_t)SK * Dd * 2);
  u16* Kb = (u16*)alloc((size_t)SK * Dd * 2);
  u16* Vt = (u16*)alloc((size_t)Dd * SK * 2);
  float* attn_out = (float*)alloc((size_t)SK * Dd * 4);
  u16* outb = (u16*)alloc((size_t)Ss * (Kk * Dd) * 2);

  // 0) zero router accumulators
  k_zero_f32<<<1, 64, 0, stream>>>(F_sum, Hh);
  k_zero_f32<<<1, 64, 0, stream>>>(P_sum, Hh);
  k_zero_f32<<<1, 64, 0, stream>>>(ent_z, 2);

  // 1) conversions
  k_cvt_bf16<<<(Ss * Ee + 255) / 256, 256, 0, stream>>>(x, xb, Ss * Ee);
  k_cvt_transpose<<<(Ee * HD + 255) / 256, 256, 0, stream>>>(Wq, WqT, Ee, HD);
  k_cvt_transpose<<<(Ee * HD + 255) / 256, 256, 0, stream>>>(Wk, WkT, Ee, HD);
  k_cvt_transpose<<<(Ee * HD + 255) / 256, 256, 0, stream>>>(Wv, WvT, Ee, HD);
  k_cvt_transpose<<<(Ee * Ee + 255) / 256, 256, 0, stream>>>(Wo, WoT, Kk * Dd, Ee);

  // 2) router
  k_router<<<Ss / 4, 128, 0, stream>>>(x, Wr, head_idx, head_w, F_sum, P_sum, ent_z);

  // 3) QKV projections (bf16 WMMA GEMMs), M=S N=HD Kd=E, 16x64 tile/wave
  dim3 gg(HD / 256, Ss / 16);
  k_gemm_bf16<<<gg, 128, 0, stream>>>(xb, WqT, q_all, Ss, HD, Ee);
  k_gemm_bf16<<<gg, 128, 0, stream>>>(xb, WkT, k_all, Ss, HD, Ee);
  k_gemm_bf16<<<gg, 128, 0, stream>>>(xb, WvT, v_all, Ss, HD, Ee);

  // 4) positions / counts / aux
  k_rank_aux<<<1, 32, 0, stream>>>(head_idx, counts_in, positions, counts_out,
                                   F_sum, P_sum, ent_z, aux_out);

  // 5) gather + RoPE
  k_gather_rope<<<SK / 4, 128, 0, stream>>>(q_all, k_all, v_all, head_idx,
                                            positions, Qb, Kb, Vt);

  // 6) flash attention (4 waves/block, async LDS staging of shared K/V blocks)
  k_flash<<<(SK / 16) / 4, 128, 0, stream>>>(Qb, Kb, Vt, head_idx, head_w, attn_out);

  // 7) output projection: (S x K*D) @ Wo -> (S x E)
  k_cvt_bf16<<<(SK * Dd + 255) / 256, 256, 0, stream>>>(attn_out, outb, SK * Dd);
  dim3 go(Ee / 256, Ss / 16);
  k_gemm_bf16<<<go, 128, 0, stream>>>(outb, WoT, out_f, Ss, Ee, Kk * Dd);
}
